// InfoNCELoss_2422361555084
// MI455X (gfx1250) — matrix operand order repaired and checked by reading, not compiled, not run on hardware
//
#include <hip/hip_runtime.h>
#include <hip/hip_bf16.h>

typedef __attribute__((ext_vector_type(16))) __bf16 v16bf;
typedef __attribute__((ext_vector_type(8)))  float  v8f;

namespace {
constexpr int kN    = 8192;   // rows in x and y
constexpr int kD    = 256;    // feature dim (full K depth)
constexpr int BM    = 128;    // block tile rows (x)
constexpr int BN    = 128;    // block tile rows (y)
constexpr int KC    = 128;    // K chunk staged in LDS (2 chunks -> 2 blocks/WGP occupancy)
constexpr int LDST  = 136;    // LDS row stride in bf16 elems (128 + 8 pad -> 4-bank/row spread)
constexpr float SCALE = 20.60992915555662f;   // log2(e) / 0.07  (exp(s/T) = exp2(s*SCALE))
constexpr float INVT  = 14.285714285714286f;  // 1 / 0.07
}

// ---- fp32 -> bf16 hi/lo split (round-to-nearest-even, pure integer ops) ----
__device__ __forceinline__ unsigned short f2bf(float f) {
    unsigned u = __float_as_uint(f);
    unsigned r = u + 0x7FFFu + ((u >> 16) & 1u);
    return (unsigned short)(r >> 16);
}
__device__ __forceinline__ float bf2f(unsigned short h) {
    return __uint_as_float(((unsigned)h) << 16);
}

union Frag { v16bf v; uint4 q[2]; };

__global__ void zero_ws(float* __restrict__ ws) {
    int i = blockIdx.x * blockDim.x + threadIdx.x;
    if (i < kN + 1) ws[i] = 0.0f;
}

// rowsum[i] += sum_j exp(x_i . y_j / T)  over this block's 128x128 tile
__global__ __launch_bounds__(256)
void infonce_rowsum(const float* __restrict__ x, const float* __restrict__ y,
                    float* __restrict__ rowsum) {
    extern __shared__ unsigned char smem_raw[];
    constexpr int TILE_BYTES = BM * LDST * 2;           // 34816 B per bf16 chunk tile
    unsigned short* xh = (unsigned short*)smem_raw;
    unsigned short* xl = (unsigned short*)(smem_raw + 1 * TILE_BYTES);
    unsigned short* yh = (unsigned short*)(smem_raw + 2 * TILE_BYTES);
    unsigned short* yl = (unsigned short*)(smem_raw + 3 * TILE_BYTES);
    float* rs          = (float*)(smem_raw + 4 * TILE_BYTES);  // [BM]

    const int tid  = threadIdx.x;
    const int lane = tid & 31;
    const int wave = tid >> 5;
    const int rowBaseX = blockIdx.x * BM;
    const int rowBaseY = blockIdx.y * BN;

    if (tid < BM) rs[tid] = 0.0f;

    // ---- wave tiling: 8 waves -> 4 (M) x 2 (N); each wave owns 32x64 = 2x4 C tiles
    const int wm = wave & 3;
    const int wn = wave >> 2;
    const int mrow = wm * 32 + (lane & 15);        // + i*16 per M tile
    const int nrow = wn * 64 + (lane & 15);        // + j*16 per N tile
    const int kA = (lane >> 4) * 8;                // A frag: K {0-7,16-23} / {8-15,24-31}
    const int kB = (lane >> 4) * 8;                // B frag: K {0-7,16-23} / {8-15,24-31}

    v8f C[2][4];
#pragma unroll
    for (int i = 0; i < 2; ++i)
#pragma unroll
        for (int j = 0; j < 4; ++j) C[i][j] = (v8f)(0.0f);

#pragma unroll 1
    for (int kc = 0; kc < kD; kc += KC) {
        // ---- stage this K chunk: global fp32 -> LDS bf16 hi/lo, 128 rows x 128 cols
        for (int idx = tid; idx < BM * (KC / 4); idx += 256) {
            const int r = idx >> 5;            // KC/4 = 32 float4 per row
            const int c = (idx & 31) << 2;
            float4 vx = *(const float4*)(x + (size_t)(rowBaseX + r) * kD + kc + c);
            float4 vy = *(const float4*)(y + (size_t)(rowBaseY + r) * kD + kc + c);
            float fx[4] = {vx.x, vx.y, vx.z, vx.w};
            float fy[4] = {vy.x, vy.y, vy.z, vy.w};
            unsigned short hx[4], lx[4], hy[4], ly[4];
#pragma unroll
            for (int k = 0; k < 4; ++k) {
                hx[k] = f2bf(fx[k]); lx[k] = f2bf(fx[k] - bf2f(hx[k]));
                hy[k] = f2bf(fy[k]); ly[k] = f2bf(fy[k] - bf2f(hy[k]));
            }
            const int off = r * LDST + c;
            *(uint2*)(xh + off) = make_uint2(hx[0] | (unsigned)hx[1] << 16, hx[2] | (unsigned)hx[3] << 16);
            *(uint2*)(xl + off) = make_uint2(lx[0] | (unsigned)lx[1] << 16, lx[2] | (unsigned)lx[3] << 16);
            *(uint2*)(yh + off) = make_uint2(hy[0] | (unsigned)hy[1] << 16, hy[2] | (unsigned)hy[3] << 16);
            *(uint2*)(yl + off) = make_uint2(ly[0] | (unsigned)ly[1] << 16, ly[2] | (unsigned)ly[3] << 16);
        }
        __syncthreads();

#pragma unroll 1
        for (int kk = 0; kk < KC; kk += 32) {
            Frag ah[2], al[2], bh[4], bl[4];
#pragma unroll
            for (int i = 0; i < 2; ++i) {
                const unsigned short* p = xh + (mrow + i * 16) * LDST + kk + kA;
                const unsigned short* q = xl + (mrow + i * 16) * LDST + kk + kA;
                ah[i].q[0] = *(const uint4*)p;        ah[i].q[1] = *(const uint4*)(p + 16);
                al[i].q[0] = *(const uint4*)q;        al[i].q[1] = *(const uint4*)(q + 16);
            }
#pragma unroll
            for (int j = 0; j < 4; ++j) {
                const unsigned short* p = yh + (nrow + j * 16) * LDST + kk + kB;
                const unsigned short* q = yl + (nrow + j * 16) * LDST + kk + kB;
                bh[j].q[0] = *(const uint4*)p;        bh[j].q[1] = *(const uint4*)(p + 16);
                bl[j].q[0] = *(const uint4*)q;        bl[j].q[1] = *(const uint4*)(q + 16);
            }
#pragma unroll
            for (int i = 0; i < 2; ++i)
#pragma unroll
                for (int j = 0; j < 4; ++j) {
                    // fp32-accurate product: hi*hi + hi*lo + lo*hi
                    C[i][j] = __builtin_amdgcn_wmma_f32_16x16x32_bf16(
                        false, ah[i].v, false, bh[j].v, (short)0, C[i][j], false, false);
                    C[i][j] = __builtin_amdgcn_wmma_f32_16x16x32_bf16(
                        false, ah[i].v, false, bl[j].v, (short)0, C[i][j], false, false);
                    C[i][j] = __builtin_amdgcn_wmma_f32_16x16x32_bf16(
                        false, al[i].v, false, bh[j].v, (short)0, C[i][j], false, false);
                }
        }
        __syncthreads();   // protect LDS before restaging next K chunk
    }

    // ---- fused epilogue: exp2(C*SCALE) via raw v_exp_f32, reduce over N, LDS rowsum
#pragma unroll
    for (int i = 0; i < 2; ++i) {
#pragma unroll
        for (int v = 0; v < 8; ++v) {
            float r = 0.0f;
#pragma unroll
            for (int j = 0; j < 4; ++j)
                r += __builtin_amdgcn_exp2f(C[i][j][v] * SCALE);
            // C layout: VGPR v, lanes 0-15 -> M=v, lanes 16-31 -> M=v+8; N = lane&15
            r += __shfl_xor(r, 8, 32);
            r += __shfl_xor(r, 4, 32);
            r += __shfl_xor(r, 2, 32);
            r += __shfl_xor(r, 1, 32);
            if ((lane & 15) == 0) {
                const int row = wm * 32 + i * 16 + v + ((lane >> 4) << 3);
                atomicAdd(&rs[row], r);
            }
        }
    }
    __syncthreads();
    if (tid < BM) unsafeAtomicAdd(&rowsum[rowBaseX + tid], rs[tid]);
}

// per-row: term = log(rowsum_i - exp(d_i/T)) - d_i/T, d_i = x_i . y_i (fp32)
__global__ __launch_bounds__(256)
void infonce_finalize_rows(const float* __restrict__ x, const float* __restrict__ y,
                           const float* __restrict__ rowsum, float* __restrict__ acc) {
    const int lane = threadIdx.x & 31;
    const int wave = threadIdx.x >> 5;
    const int rowBase = blockIdx.x * 128 + wave * 16;
    float local = 0.0f;
    for (int rr = 0; rr < 16; ++rr) {
        const int r = rowBase + rr;
        float d = 0.0f;
#pragma unroll
        for (int c = 0; c < kD; c += 32)
            d = fmaf(x[(size_t)r * kD + c + lane], y[(size_t)r * kD + c + lane], d);
        d += __shfl_xor(d, 16, 32);
        d += __shfl_xor(d, 8, 32);
        d += __shfl_xor(d, 4, 32);
        d += __shfl_xor(d, 2, 32);
        d += __shfl_xor(d, 1, 32);
        if (lane == 0) {
            const float neg = rowsum[r] - __builtin_amdgcn_exp2f(d * SCALE);
            local += logf(neg) - d * INVT;
        }
    }
    if (lane == 0) unsafeAtomicAdd(acc, local);
}

__global__ void infonce_write(const float* __restrict__ acc, float* __restrict__ out) {
    out[0] = acc[0] * (1.0f / kN);
}

extern "C" void kernel_launch(void* const* d_in, const int* in_sizes, int n_in,
                              void* d_out, int out_size, void* d_ws, size_t ws_size,
                              hipStream_t stream) {
    const float* x = (const float*)d_in[0];
    const float* y = (const float*)d_in[1];
    float* ws  = (float*)d_ws;     // [0..8191] rowsum, [8192] loss accumulator
    float* out = (float*)d_out;

    // ~136.5 KB dynamic LDS -> 2 workgroups per 320 KB WGP -> 4 waves/SIMD
    constexpr size_t SMEM = (size_t)4 * BM * LDST * 2 + BM * sizeof(float);
    hipFuncSetAttribute((const void*)infonce_rowsum,
                        hipFuncAttributeMaxDynamicSharedMemorySize, (int)SMEM);

    zero_ws<<<(kN + 1 + 255) / 256, 256, 0, stream>>>(ws);
    dim3 grid(kN / BM, kN / BN);   // 64 x 64 = 4096 blocks
    infonce_rowsum<<<grid, 256, SMEM, stream>>>(x, y, ws);
    infonce_finalize_rows<<<kN / 128, 256, 0, stream>>>(x, y, ws, ws + kN);
    infonce_write<<<1, 1, 0, stream>>>(ws + kN, out);
}
